// TransformerBlock_80238579024584
// MI455X (gfx1250) — matrix operand order repaired and checked by reading, not compiled, not run on hardware
//
#include <hip/hip_runtime.h>
#include <hip/hip_bf16.h>
#include <math.h>

typedef __attribute__((ext_vector_type(16))) __bf16 v16bf;
typedef __attribute__((ext_vector_type(8)))  float  v8f;

union FragAB { v16bf v; unsigned int u[8]; };

// Pack two f32 into one dword of bf16x2 with round-to-nearest-even.
__device__ __forceinline__ unsigned int pack_bf16x2(float a, float b) {
  union { float f; unsigned int u; } x, y;
  x.f = a; y.f = b;
  unsigned int xu = x.u + (0x7fffu + ((x.u >> 16) & 1u));
  unsigned int yu = y.u + (0x7fffu + ((y.u >> 16) & 1u));
  return (xu >> 16) | (yu & 0xffff0000u);
}

// ---------------------------------------------------------------------------
// Weight cast+transpose (runs once per layer call, tiny):
// W[K][N] f32 -> Wt[N][K] bf16, so GEMM B-tiles are contiguous over K.
// ---------------------------------------------------------------------------
__global__ __launch_bounds__(256) void cast_transpose_k(
    const float* __restrict__ W, __hip_bfloat16* __restrict__ Wt, int K, int N)
{
  __shared__ float tile[32][33];
  const int tx = threadIdx.x & 31, ty = threadIdx.x >> 5;  // ty 0..7
  const int k0 = blockIdx.y * 32, n0 = blockIdx.x * 32;
  #pragma unroll
  for (int r = 0; r < 4; ++r)
    tile[ty + 8 * r][tx] = W[(size_t)(k0 + ty + 8 * r) * N + n0 + tx];
  __syncthreads();
  #pragma unroll
  for (int r = 0; r < 4; ++r)
    Wt[(size_t)(n0 + ty + 8 * r) * K + k0 + tx] = __float2bfloat16(tile[tx][ty + 8 * r]);
}

// ---------------------------------------------------------------------------
// LayerNorm over last dim, one block per row; f32 in, bf16 out (packed dwords).
// ---------------------------------------------------------------------------
__global__ __launch_bounds__(256) void layernorm_bf16_k(
    const float* __restrict__ x, const float* __restrict__ g,
    const float* __restrict__ b, __hip_bfloat16* __restrict__ y, int D)
{
  __shared__ float red[256];
  __shared__ float red2[256];
  const int row = blockIdx.x;
  const float* xr = x + (size_t)row * D;
  float s = 0.f, s2 = 0.f;
  for (int i = threadIdx.x; i < D; i += 256) { float v = xr[i]; s += v; s2 += v * v; }
  red[threadIdx.x] = s; red2[threadIdx.x] = s2;
  __syncthreads();
  for (int off = 128; off > 0; off >>= 1) {
    if ((int)threadIdx.x < off) {
      red[threadIdx.x]  += red[threadIdx.x + off];
      red2[threadIdx.x] += red2[threadIdx.x + off];
    }
    __syncthreads();
  }
  const float mu  = red[0] / (float)D;
  const float var = red2[0] / (float)D - mu * mu;
  const float rs  = rsqrtf(var + 1e-5f);
  unsigned int* yu = (unsigned int*)(y + (size_t)row * D);
  for (int i = threadIdx.x * 2; i < D; i += 512) {
    const float v0 = (xr[i]     - mu) * rs * g[i]     + b[i];
    const float v1 = (xr[i + 1] - mu) * rs * g[i + 1] + b[i + 1];
    yu[i >> 1] = pack_bf16x2(v0, v1);
  }
}

// ---------------------------------------------------------------------------
// bf16 WMMA GEMM: C[M,N] = act(A[M,K] @ B[K,N] + bias) (+ R)
// A bf16 row-major, Bt bf16 = B transposed ([N][K], K contiguous).
// Block 256 thr = 8 waves; tile 128x128, K-step 64 (2 WMMA k-steps/barrier).
// Staging is pure uint4 (b128) copies -> no conversion VALU in the hot loop.
// ACT: 0=none, 1=exact GELU.  RES: add f32 R.  OUTBF: store bf16 else f32.
// ---------------------------------------------------------------------------
constexpr int BM = 128, BN = 128, BK = 64;

template<int ACT, bool RES, bool OUTBF>
__global__ __launch_bounds__(256) void gemm2(
    const __hip_bfloat16* __restrict__ A, const __hip_bfloat16* __restrict__ Bt,
    const float* __restrict__ bias, const float* __restrict__ R,
    void* __restrict__ Cout, int M, int K, int N)
{
  // 64 bf16 = 32 dwords data + pad to 40 dwords (160B, 16B-aligned rows).
  __shared__ alignas(16) unsigned int As[BM][40];
  __shared__ alignas(16) unsigned int Bs[BN][40];

  const int tid  = threadIdx.x;
  const int wid  = tid >> 5;
  const int lane = tid & 31;
  const int half = lane >> 4;
  const int l16  = lane & 15;
  const int wr   = wid & 3;    // 4 row-groups of 32
  const int wc   = wid >> 2;   // 2 col-groups of 64
  const int rowBase = blockIdx.y * BM;
  const int colBase = blockIdx.x * BN;
  const unsigned short* Au = (const unsigned short*)A;
  const unsigned short* Bu = (const unsigned short*)Bt;

  v8f acc[2][4] = {};

  for (int k0 = 0; k0 < K; k0 += BK) {
    // Stage A tile (128 x 64 bf16): 1024 uint4 chunks, 4 per thread.
    #pragma unroll
    for (int t = 0; t < 4; ++t) {
      const int idx = t * 256 + tid;
      const int ch = idx & 7, row = idx >> 3;
      const uint4 d = *(const uint4*)(Au + (size_t)(rowBase + row) * K + k0 + ch * 8);
      *(uint4*)&As[row][ch * 4] = d;
    }
    // Stage Bt tile (128 cols x 64 bf16 along K): same pattern.
    #pragma unroll
    for (int t = 0; t < 4; ++t) {
      const int idx = t * 256 + tid;
      const int ch = idx & 7, col = idx >> 3;
      const uint4 d = *(const uint4*)(Bu + (size_t)(colBase + col) * K + k0 + ch * 8);
      *(uint4*)&Bs[col][ch * 4] = d;
    }
    if (k0 + BK < K) {  // hint next tiles (global_prefetch_b8)
      __builtin_prefetch(Au + (size_t)(rowBase + (tid >> 1)) * K + k0 + BK, 0, 1);
      __builtin_prefetch(Bu + (size_t)(colBase + (tid >> 1)) * K + k0 + BK, 0, 1);
    }
    __syncthreads();

    #pragma unroll
    for (int kk = 0; kk < 2; ++kk) {
      FragAB afr[2], bfr[4];
      #pragma unroll
      for (int tm = 0; tm < 2; ++tm) {
        const int r = wr * 32 + tm * 16 + l16;
        #pragma unroll
        for (int i = 0; i < 4; ++i) {
          afr[tm].u[i]     = As[r][kk * 16 + half * 4 + i];       // K = half*8 + 2i,+1
          afr[tm].u[4 + i] = As[r][kk * 16 + 8 + half * 4 + i];   // K = 16 + half*8 + 2i,+1
        }
      }
      #pragma unroll
      for (int tn = 0; tn < 4; ++tn) {
        const int c = wc * 64 + tn * 16 + l16;
        #pragma unroll
        for (int i = 0; i < 4; ++i) {
          bfr[tn].u[i]     = Bs[c][kk * 16 + half * 4 + i];
          bfr[tn].u[4 + i] = Bs[c][kk * 16 + 8 + half * 4 + i];
        }
      }
      #pragma unroll
      for (int tm = 0; tm < 2; ++tm)
        #pragma unroll
        for (int tn = 0; tn < 4; ++tn)
          acc[tm][tn] = __builtin_amdgcn_wmma_f32_16x16x32_bf16(
              false, afr[tm].v, false, bfr[tn].v, (short)0, acc[tm][tn], false, false);
    }
    __syncthreads();
  }

  // Epilogue: bias, activation, residual, store.
  #pragma unroll
  for (int tm = 0; tm < 2; ++tm) {
    #pragma unroll
    for (int tn = 0; tn < 4; ++tn) {
      const int col = colBase + wc * 64 + tn * 16 + l16;
      const float bval = bias ? bias[col] : 0.f;
      #pragma unroll
      for (int r = 0; r < 8; ++r) {
        const int row = rowBase + wr * 32 + tm * 16 + half * 8 + r;
        float v = acc[tm][tn][r] + bval;
        if (ACT == 1) v = 0.5f * v * (1.f + erff(v * 0.70710678118654752f));
        if (RES) v += R[(size_t)row * N + col];
        if (OUTBF) ((__hip_bfloat16*)Cout)[(size_t)row * N + col] = __float2bfloat16(v);
        else       ((float*)Cout)[(size_t)row * N + col] = v;
      }
    }
  }
}

// ---------------------------------------------------------------------------
// Flash attention, all-bf16 I/O. One 128-thr block per (b, h, 64-row Q block).
// qkv: bf16 [B*N, 3*D], col = which*D + h*64 + hd.  O: bf16 [B*N, D].
// ---------------------------------------------------------------------------
__global__ __launch_bounds__(128) void attn2_k(
    const __hip_bfloat16* __restrict__ qkv, __hip_bfloat16* __restrict__ O)
{
  constexpr int N = 1024, D = 1024, HD = 64;
  constexpr float SCALE = 0.125f;  // 1/sqrt(64)

  __shared__ alignas(16) unsigned int Qs[64][36];  // [qrow][HD pair]  (A layout)
  __shared__ alignas(16) unsigned int Ks[64][36];  // [krow][HD pair]  (B^T layout)
  __shared__ alignas(16) unsigned int Vt[64][36];  // [hd][seq pair]   (B^T layout)
  __shared__ alignas(16) unsigned int Ps[64][36];  // [qrow][seq pair] (A layout)
  __shared__ float Ss[64][66];
  __shared__ float mrow[64], lrow[64], arow[64];

  const int tid = threadIdx.x;
  const int w = tid >> 5, lane = tid & 31, half = lane >> 4, l16 = lane & 15;
  int bid = blockIdx.x;
  const int qb = bid & 15; bid >>= 4;
  const int h  = bid & 15; bid >>= 4;
  const int b  = bid;
  const unsigned short* qu = (const unsigned short*)qkv;

  // Stage Q block (64 x 64 bf16) once: pure b128 copies.
  #pragma unroll
  for (int t = 0; t < 4; ++t) {
    const int idx = t * 128 + tid;
    const int ch = idx & 7, row = idx >> 3;
    const uint4 d = *(const uint4*)(qu + (size_t)(b * N + qb * 64 + row) * 3 * D + h * HD + ch * 8);
    *(uint4*)&Qs[row][ch * 4] = d;
  }
  if (tid < 64) { mrow[tid] = -3.0e38f; lrow[tid] = 0.f; }
  v8f acco[4] = {};
  __syncthreads();

  for (int kb = 0; kb < N / 64; ++kb) {
    // Stage K block (row-major == B^T layout for QK^T).
    #pragma unroll
    for (int t = 0; t < 4; ++t) {
      const int idx = t * 128 + tid;
      const int ch = idx & 7, row = idx >> 3;
      const uint4 d = *(const uint4*)(qu + (size_t)(b * N + kb * 64 + row) * 3 * D + D + h * HD + ch * 8);
      *(uint4*)&Ks[row][ch * 4] = d;
    }
    // Stage V transposed: Vt[hd][seqpair] (genuine transpose: u16 gathers).
    #pragma unroll
    for (int t = 0; t < 16; ++t) {
      const int idx = t * 128 + tid;
      const int hd = idx & 63, sp = idx >> 6;
      const unsigned v0 = qu[(size_t)(b * N + kb * 64 + 2 * sp)     * 3 * D + 2 * D + h * HD + hd];
      const unsigned v1 = qu[(size_t)(b * N + kb * 64 + 2 * sp + 1) * 3 * D + 2 * D + h * HD + hd];
      Vt[hd][sp] = v0 | (v1 << 16);
    }
    __syncthreads();

    // S = Q @ K^T for this wave's 16-row strip (2 k-steps over HD=64).
    v8f accs[4] = {};
    #pragma unroll
    for (int kk = 0; kk < 2; ++kk) {
      FragAB qa;
      const int r = w * 16 + l16;
      #pragma unroll
      for (int i = 0; i < 4; ++i) {
        qa.u[i]     = Qs[r][kk * 16 + half * 4 + i];
        qa.u[4 + i] = Qs[r][kk * 16 + 8 + half * 4 + i];
      }
      #pragma unroll
      for (int tn = 0; tn < 4; ++tn) {
        FragAB kf;
        const int c = tn * 16 + l16;
        #pragma unroll
        for (int i = 0; i < 4; ++i) {
          kf.u[i]     = Ks[c][kk * 16 + half * 4 + i];
          kf.u[4 + i] = Ks[c][kk * 16 + 8 + half * 4 + i];
        }
        accs[tn] = __builtin_amdgcn_wmma_f32_16x16x32_bf16(
            false, qa.v, false, kf.v, (short)0, accs[tn], false, false);
      }
    }
    #pragma unroll
    for (int tn = 0; tn < 4; ++tn)
      #pragma unroll
      for (int r = 0; r < 8; ++r)
        Ss[w * 16 + half * 8 + r][tn * 16 + l16] = accs[tn][r] * SCALE;
    __syncthreads();

    // Online softmax: one thread per row.
    if (tid < 64) {
      const int row = tid;
      const float mold = mrow[row];
      float mx = mold;
      #pragma unroll 8
      for (int j = 0; j < 64; ++j) mx = fmaxf(mx, Ss[row][j]);
      const float alpha = __expf(mold - mx);
      float l = lrow[row] * alpha;
      #pragma unroll 4
      for (int jp = 0; jp < 32; ++jp) {
        const float p0 = __expf(Ss[row][2 * jp]     - mx);
        const float p1 = __expf(Ss[row][2 * jp + 1] - mx);
        l += p0 + p1;
        Ps[row][jp] = pack_bf16x2(p0, p1);
      }
      mrow[row] = mx; lrow[row] = l; arow[row] = alpha;
    }
    __syncthreads();

    // Rescale running O, then O += P @ V.
    #pragma unroll
    for (int tn = 0; tn < 4; ++tn)
      #pragma unroll
      for (int r = 0; r < 8; ++r)
        acco[tn][r] *= arow[w * 16 + half * 8 + r];

    #pragma unroll
    for (int kk = 0; kk < 2; ++kk) {
      FragAB pa;
      const int r = w * 16 + l16;
      #pragma unroll
      for (int i = 0; i < 4; ++i) {
        pa.u[i]     = Ps[r][kk * 16 + half * 4 + i];
        pa.u[4 + i] = Ps[r][kk * 16 + 8 + half * 4 + i];
      }
      #pragma unroll
      for (int tn = 0; tn < 4; ++tn) {
        FragAB vf;
        const int c = tn * 16 + l16;
        #pragma unroll
        for (int i = 0; i < 4; ++i) {
          vf.u[i]     = Vt[c][kk * 16 + half * 4 + i];
          vf.u[4 + i] = Vt[c][kk * 16 + 8 + half * 4 + i];
        }
        acco[tn] = __builtin_amdgcn_wmma_f32_16x16x32_bf16(
            false, pa.v, false, vf.v, (short)0, acco[tn], false, false);
      }
    }
    __syncthreads();
  }

  // Normalize and store O (bf16, head-concatenated [B*N, D]).
  #pragma unroll
  for (int tn = 0; tn < 4; ++tn)
    #pragma unroll
    for (int r = 0; r < 8; ++r) {
      const int rl = w * 16 + half * 8 + r;
      O[(size_t)(b * N + qb * 64 + rl) * D + h * HD + tn * 16 + l16] =
          __float2bfloat16(acco[tn][r] / lrow[rl]);
    }
}

// ---------------------------------------------------------------------------
// Driver
// ---------------------------------------------------------------------------
extern "C" void kernel_launch(void* const* d_in, const int* in_sizes, int n_in,
                              void* d_out, int out_size, void* d_ws, size_t ws_size,
                              hipStream_t stream) {
  (void)in_sizes; (void)n_in; (void)out_size; (void)ws_size;
  constexpr int Bc = 4, Nc = 1024, Dc = 1024, FFc = 2048;
  constexpr int Mr = Bc * Nc;  // 4096 rows

  const float* x     = (const float*)d_in[0];
  const float* ln1_g = (const float*)d_in[1];
  const float* ln1_b = (const float*)d_in[2];
  const float* qkv_w = (const float*)d_in[3];
  const float* qkv_b = (const float*)d_in[4];
  const float* out_w = (const float*)d_in[5];
  const float* out_b = (const float*)d_in[6];
  const float* ln2_g = (const float*)d_in[7];
  const float* ln2_b = (const float*)d_in[8];
  const float* w1    = (const float*)d_in[9];
  const float* b1    = (const float*)d_in[10];
  const float* w2    = (const float*)d_in[11];
  const float* b2    = (const float*)d_in[12];
  float* out = (float*)d_out;

  char* w = (char*)d_ws;
  __hip_bfloat16* xnb  = (__hip_bfloat16*)w; w += (size_t)Mr * Dc * 2;       // ln out (reused)
  __hip_bfloat16* qkvb = (__hip_bfloat16*)w; w += (size_t)Mr * 3 * Dc * 2;   // qkv / ffn hidden
  __hip_bfloat16* ao   = (__hip_bfloat16*)w; w += (size_t)Mr * Dc * 2;       // attention out
  float*          y    = (float*)w;          w += (size_t)Mr * Dc * 4;       // residual-1 (f32)
  __hip_bfloat16* Wqt  = (__hip_bfloat16*)w; w += (size_t)3 * Dc * Dc * 2;   // qkv_w^T
  __hip_bfloat16* Wot  = (__hip_bfloat16*)w; w += (size_t)Dc * Dc * 2;       // out_w^T
  __hip_bfloat16* W1t  = (__hip_bfloat16*)w; w += (size_t)FFc * Dc * 2;      // w1^T
  __hip_bfloat16* W2t  = (__hip_bfloat16*)w; w += (size_t)Dc * FFc * 2;      // w2^T
  __hip_bfloat16* h1b  = qkvb;                                               // reuse

  // One-time weight cast+transpose to bf16 [N][K].
  cast_transpose_k<<<dim3(3 * Dc / 32, Dc / 32), 256, 0, stream>>>(qkv_w, Wqt, Dc, 3 * Dc);
  cast_transpose_k<<<dim3(Dc / 32, Dc / 32),     256, 0, stream>>>(out_w, Wot, Dc, Dc);
  cast_transpose_k<<<dim3(FFc / 32, Dc / 32),    256, 0, stream>>>(w1, W1t, Dc, FFc);
  cast_transpose_k<<<dim3(Dc / 32, FFc / 32),    256, 0, stream>>>(w2, W2t, FFc, Dc);

  // x -> ln1 -> qkv -> attention -> out-proj(+x) -> ln2 -> ffn1(gelu) -> ffn2(+y)
  layernorm_bf16_k<<<Mr, 256, 0, stream>>>(x, ln1_g, ln1_b, xnb, Dc);
  gemm2<0, false, true><<<dim3(3 * Dc / BN, Mr / BM), 256, 0, stream>>>(
      xnb, Wqt, qkv_b, nullptr, qkvb, Mr, Dc, 3 * Dc);
  attn2_k<<<Bc * 16 * (Nc / 64), 128, 0, stream>>>(qkvb, ao);
  gemm2<0, true, false><<<dim3(Dc / BN, Mr / BM), 256, 0, stream>>>(
      ao, Wot, out_b, x, y, Mr, Dc, Dc);
  layernorm_bf16_k<<<Mr, 256, 0, stream>>>(y, ln2_g, ln2_b, xnb, Dc);
  gemm2<1, false, true><<<dim3(FFc / BN, Mr / BM), 256, 0, stream>>>(
      xnb, W1t, b1, nullptr, h1b, Mr, Dc, FFc);
  gemm2<0, true, false><<<dim3(Dc / BN, Mr / BM), 256, 0, stream>>>(
      h1b, W2t, b2, y, out, Mr, FFc, Dc);
}
